// VanillaRNN_7430293422447
// MI455X (gfx1250) — matrix operand order repaired and checked by reading, not compiled, not run on hardware
//
#include <hip/hip_runtime.h>

// ---------------------------------------------------------------------------
// VanillaRNN on MI455X (gfx1250, wave32, WMMA).
// All GEMMs use v_wmma_f32_16x16x32_bf16 (fp32 accumulate). Weights converted
// fp32->bf16 once into workspace. Recurrence runs as a persistent grid with
// LDS-resident W_hh column slices + atomic-counter grid sync. Fragment loads
// are ds_load_b128; A-tile staging in the big GEMM and the per-step h
// broadcast use GLOBAL_LOAD_ASYNC_TO_LDS_B128 tracked on ASYNCcnt.
// ---------------------------------------------------------------------------

typedef __attribute__((ext_vector_type(16))) __bf16  v16bf;
typedef __attribute__((ext_vector_type(8)))  float   v8f;

union ABfrag { v16bf v; unsigned int u[8]; uint4 q[2]; };
union U16x16 { uint4 q[2]; unsigned short h[16]; };

#define RNN_B   8
#define RNN_T   256
#define RNN_E   512
#define RNN_H   1024
#define RNN_V   32000
#define RNN_M   (RNN_B * RNN_T)   // 2048 flattened (b,t) rows
#define REC_NWG 64                // recurrence workgroups (16 cols each)

__device__ __forceinline__ unsigned short f32_to_bf16_rne(float f) {
  unsigned int u = __float_as_uint(f);
  unsigned int lsb = (u >> 16) & 1u;
  u += 0x7fffu + lsb;               // round-to-nearest-even
  return (unsigned short)(u >> 16);
}

__device__ __forceinline__ unsigned int pack_bf16x2(float lo, float hi) {
  return (unsigned int)f32_to_bf16_rne(lo) |
         ((unsigned int)f32_to_bf16_rne(hi) << 16);
}

__device__ __forceinline__ v8f wmma_bf16(const ABfrag& a, const ABfrag& b, v8f c) {
  // 8 args: (neg_a, A, neg_b, B, c_mod, C, reuse_a, reuse_b)
  return __builtin_amdgcn_wmma_f32_16x16x32_bf16(false, a.v, false, b.v,
                                                 (short)0, c, false, false);
}

// ---- async global -> LDS (GV mode, VDST = LDS byte address VGPR) ----------
__device__ __forceinline__ void async_load_b128(const void* gsrc, void* ldst) {
  unsigned lds = (unsigned)(uintptr_t)ldst;   // LDS aperture: addr[31:0]
  asm volatile("global_load_async_to_lds_b128 %0, %1, off"
               :: "v"(lds), "v"(gsrc) : "memory");
}
__device__ __forceinline__ void wait_async0() {
  asm volatile("s_wait_asynccnt 0x0" ::: "memory");
}

// ---------------------------------------------------------------------------
// fp32 -> bf16 bulk conversion (grid-stride)
// ---------------------------------------------------------------------------
__global__ void cvt_f32_bf16_kernel(const float* __restrict__ src,
                                    unsigned short* __restrict__ dst, int n) {
  int i = blockIdx.x * blockDim.x + threadIdx.x;
  int stride = gridDim.x * blockDim.x;
  for (; i < n; i += stride) dst[i] = f32_to_bf16_rne(src[i]);
}

// ---------------------------------------------------------------------------
// Init: zero the h exchange buffer and the grid-sync counter (every call,
// since the harness graph-replays kernel_launch).
// ---------------------------------------------------------------------------
__global__ void rnn_init_kernel(unsigned short* __restrict__ hglob,
                                int* __restrict__ counter) {
  int i = blockIdx.x * blockDim.x + threadIdx.x;
  if (i < RNN_B * RNN_H) hglob[i] = 0;
  if (i == 0) *counter = 0;
}

// ---------------------------------------------------------------------------
// Generic bf16 WMMA GEMM:  C[M,N] = A[M,K] * B[K,N] + bias[N]   (C fp32)
//   GATHER=true : A row m is embedding[idx[m]] in fp32, converted on stage-in
//                 (register path; conversion needs VALU).
//   GATHER=false: A is bf16 row-major -> staged via async-to-LDS b128 DMA.
// 256 threads = 8 waves (2x4); block tile 128x128, BK=32; per-wave tile
// 64x32 = 4x2 sub-tiles -> 8 wmma per K-step. Double-buffered LDS staging:
// async A copies into buf 1-p are issued right after the barrier that
// publishes buf p and completed (s_wait_asynccnt 0) before the next barrier.
// M, N, K must be multiples of 128/128/32 (true for all call sites).
// ---------------------------------------------------------------------------
template <bool GATHER>
__global__ __launch_bounds__(256)
void gemm_bf16_wmma_kernel(const void* __restrict__ Aptr,
                           const int* __restrict__ idx,
                           const unsigned short* __restrict__ Bw,
                           const float* __restrict__ bias,
                           float* __restrict__ C, int M, int N, int K) {
  __shared__ unsigned short As[2][128 * 32];   // [row][k]            2 x 8 KB
  __shared__ unsigned short Bs[2][128 * 32];   // [col][k] transposed 2 x 8 KB

  const int m0   = blockIdx.x * 128;
  const int n0   = blockIdx.y * 128;
  const int tid  = threadIdx.x;
  const int wave = tid >> 5;
  const int lane = tid & 31;
  const int wm   = wave >> 2;      // 0..1 (M)
  const int wn   = wave & 3;       // 0..3 (N)
  const int lrow = lane & 15;
  const int g    = lane >> 4;      // lane group (0: lanes 0-15, 1: 16-31)

  const v8f vz = {0.f, 0.f, 0.f, 0.f, 0.f, 0.f, 0.f, 0.f};
  v8f acc[4][2];
#pragma unroll
  for (int mi = 0; mi < 4; mi++)
#pragma unroll
    for (int ni = 0; ni < 2; ni++) acc[mi][ni] = vz;

  // staging assignments (each thread: 16 bf16 of A, 16 bf16 of B)
  const int arow = tid >> 1;            // 0..127
  const int aseg = (tid & 1) * 16;      // 0 or 16
  const int bkl  = tid >> 3;            // 0..31  (k within tile)
  const int bns  = (tid & 7) * 16;      // 0..112 (n segment)

  // ---- loaders ----
  auto load_a_regs = [&](int k0, uint4 qa[2]) {   // gather + fp32->bf16
    const float4* s = (const float4*)((const float*)Aptr +
                      (size_t)idx[m0 + arow] * K + k0 + aseg);
    float4 f0 = s[0], f1 = s[1], f2 = s[2], f3 = s[3];
    qa[0].x = pack_bf16x2(f0.x, f0.y); qa[0].y = pack_bf16x2(f0.z, f0.w);
    qa[0].z = pack_bf16x2(f1.x, f1.y); qa[0].w = pack_bf16x2(f1.z, f1.w);
    qa[1].x = pack_bf16x2(f2.x, f2.y); qa[1].y = pack_bf16x2(f2.z, f2.w);
    qa[1].z = pack_bf16x2(f3.x, f3.y); qa[1].w = pack_bf16x2(f3.z, f3.w);
  };
  auto store_a = [&](int p, const uint4 qa[2]) {
    *(uint4*)&As[p][arow * 32 + aseg]     = qa[0];
    *(uint4*)&As[p][arow * 32 + aseg + 8] = qa[1];
  };
  auto issue_async_a = [&](int k0, int p) {       // bf16 A: pure DMA
    const unsigned short* src = (const unsigned short*)Aptr +
                                (size_t)(m0 + arow) * K + k0 + aseg;
    async_load_b128(src,     &As[p][arow * 32 + aseg]);
    async_load_b128(src + 8, &As[p][arow * 32 + aseg + 8]);
  };
  auto load_b = [&](int k0, U16x16& ub) {
    const uint4* s = (const uint4*)(Bw + (size_t)(k0 + bkl) * N + n0 + bns);
    ub.q[0] = s[0]; ub.q[1] = s[1];
  };
  auto store_b = [&](int p, const U16x16& ub) {   // transpose scatter
#pragma unroll
    for (int i = 0; i < 16; i++) Bs[p][(bns + i) * 32 + bkl] = ub.h[i];
  };

  // prologue: stage tile 0 into buffer 0
  {
    U16x16 ub; load_b(0, ub); store_b(0, ub);
    if (GATHER) { uint4 qa[2]; load_a_regs(0, qa); store_a(0, qa); }
    else        issue_async_a(0, 0);
  }

  const int nk = K / 32;
  int p = 0;
  for (int kk = 0; kk < nk; kk++) {
    if (!GATHER) wait_async0();   // my async stores into buf p are done
    __syncthreads();              // publish buf p to all waves

    uint4 nqa[2]; U16x16 nub;
    if (kk + 1 < nk) {            // overlap next tile fetch with math
      load_b((kk + 1) * 32, nub);
      if (GATHER) load_a_regs((kk + 1) * 32, nqa);
      else        issue_async_a((kk + 1) * 32, p ^ 1);
      if (kk + 2 < nk)            // hint k+2 B panel -> global_prefetch
        __builtin_prefetch(Bw + (size_t)((kk + 2) * 32 + bkl) * N + n0 + bns,
                           0, 1);
    }

    // ---- fragments (two ds_load_b128 each, per ISA VGPR layout) ----
    const uint4* As4 = (const uint4*)As[p];  // 4 uint4 per 32-half row
    const uint4* Bs4 = (const uint4*)Bs[p];
    ABfrag a[4], b[2];
#pragma unroll
    for (int mi = 0; mi < 4; mi++) {
      const int rb4 = (wm * 64 + mi * 16 + lrow) * 4;
      a[mi].q[0] = As4[rb4 + g];           // dwords g*4 .. g*4+3
      a[mi].q[1] = As4[rb4 + 2 + g];       // dwords 8+g*4 ..
    }
#pragma unroll
    for (int ni = 0; ni < 2; ni++) {
      const int cb4 = (wn * 32 + ni * 16 + lrow) * 4;
      b[ni].q[0] = Bs4[cb4 + g];           // dwords g*4 .. g*4+3
      b[ni].q[1] = Bs4[cb4 + 2 + g];       // dwords 8+g*4 ..
    }
#pragma unroll
    for (int mi = 0; mi < 4; mi++)
#pragma unroll
      for (int ni = 0; ni < 2; ni++)
        acc[mi][ni] = wmma_bf16(a[mi], b[ni], acc[mi][ni]);

    if (kk + 1 < nk) {
      store_b(p ^ 1, nub);                 // fill other buffer (reg path)
      if (GATHER) store_a(p ^ 1, nqa);
    }
    p ^= 1;
  }

  // ---- epilogue: C layout row = g*8 + r, col = lane&15 ----
#pragma unroll
  for (int mi = 0; mi < 4; mi++) {
#pragma unroll
    for (int ni = 0; ni < 2; ni++) {
      const int gr = m0 + wm * 64 + mi * 16 + g * 8;
      const int gc = n0 + wn * 32 + ni * 16 + lrow;
      const float bv = bias[gc];
#pragma unroll
      for (int r = 0; r < 8; r++)
        C[(size_t)(gr + r) * N + gc] = acc[mi][ni][r] + bv;
    }
  }
}

// ---------------------------------------------------------------------------
// Persistent recurrence: h_t = tanh(xh_t + h_{t-1} @ W_hh), 256 steps.
// 64 WGs x 64 threads (2 waves). WG w owns columns [16w, 16w+16) with its
// W_hh slice (1024x16 bf16 = 32 KB) resident in LDS. h is kept as a full
// padded 16x1024 LDS tile (rows 8..15 zeroed once) so fragment loads are
// unconditional ds_load_b128. Grid sync via fence + atomic counter; the
// per-step 16 KB h broadcast is pulled global->LDS with async b128 DMA.
// Dynamic LDS: 32 KB (Whh) + 32 KB (h padded) + 1 KB (reduce) = 65 KB.
// ---------------------------------------------------------------------------
__global__ __launch_bounds__(64)
void rnn_recurrence_kernel(const unsigned short* __restrict__ Whh,  // [H][H] bf16
                           const float* __restrict__ xh,            // [B,T,H] f32
                           unsigned short* __restrict__ hs,         // [B,T,H] bf16 out
                           unsigned short* __restrict__ hglob,      // [B][H] bf16
                           int* __restrict__ counter) {
  extern __shared__ unsigned char smem[];
  unsigned short* whh_s = (unsigned short*)smem;              // [16 cols][1024 k]
  unsigned short* h_s   = whh_s + 16 * RNN_H;                 // [16 rows][1024 k]
  float*          red   = (float*)(h_s + 16 * RNN_H);         // [32 lanes][8]

  const int wg   = blockIdx.x;        // 0..63
  const int n0   = wg * 16;           // owned column base
  const int tid  = threadIdx.x;       // 0..63
  const int wv   = tid >> 5;          // wave 0/1 -> K halves
  const int lane = tid & 31;
  const int lrow = lane & 15;
  const int g    = lane >> 4;

  // stage W_hh slice, transposed: whh_s[col][k] (16 lanes read 16 adjacent cols)
  {
    const int col = tid & 15;
    const int kb  = (tid >> 4) * 256;
    for (int k = 0; k < 256; k++)
      whh_s[col * RNN_H + kb + k] = Whh[(size_t)(kb + k) * RNN_H + n0 + col];
  }
  // zero the full padded h tile (rows 8..15 stay zero forever)
  {
    unsigned int* ptr = (unsigned int*)h_s;
    for (int i = tid; i < 16 * RNN_H / 2; i += 64) ptr[i] = 0u;
  }
  __syncthreads();

  const uint4* h4 = (const uint4*)h_s;    // 128 uint4 per row
  const uint4* w4 = (const uint4*)whh_s;  // 128 uint4 per col
  const v8f vz = {0.f, 0.f, 0.f, 0.f, 0.f, 0.f, 0.f, 0.f};

  for (int t = 0; t < RNN_T; t++) {
    // ---- GEMM: each wave covers half of K (512) for the 16x16 tile ----
    v8f acc = vz;
    const int kq = wv * 128;            // uint4 base: 0 or 128 (128 = one full 1024-k row)
#pragma unroll
    for (int kk = 0; kk < 16; kk++) {
      const int kb4  = kq + kk * 4;     // 32 halfs = 4 uint4 per chunk
      const int abase = lrow * 128 + kb4;
      ABfrag a, b;
      a.q[0] = h4[abase + g];
      a.q[1] = h4[abase + 2 + g];
      b.q[0] = w4[abase + g];
      b.q[1] = w4[abase + 2 + g];
      acc = wmma_bf16(a, b, acc);
    }
    // ---- cross-wave K reduction through LDS ----
    if (wv == 1) {
#pragma unroll
      for (int r = 0; r < 8; r++) red[lane * 8 + r] = acc[r];
    }
    __syncthreads();
    if (wv == 0) {
#pragma unroll
      for (int r = 0; r < 8; r++) acc[r] += red[lane * 8 + r];
      // epilogue: rows g*8+r (only 0..7 are real batches), col = n0+lrow
      const int col = n0 + lrow;
#pragma unroll
      for (int r = 0; r < 8; r++) {
        const int m = g * 8 + r;
        if (m < RNN_B) {
          const size_t bt = (size_t)m * RNN_T + t;
          const float v = acc[r] + xh[bt * RNN_H + col];
          const unsigned short hb = f32_to_bf16_rne(tanhf(v));
          hs[bt * RNN_H + col]   = hb;   // stash for output projection
          hglob[m * RNN_H + col] = hb;   // broadcast to all WGs
        }
      }
    }
    // ---- grid-wide sync (monotonic counter, no reset between steps) ----
    __threadfence();
    __syncthreads();
    if (tid == 0) {
      __hip_atomic_fetch_add(counter, 1, __ATOMIC_ACQ_REL,
                             __HIP_MEMORY_SCOPE_AGENT);
      const int target = REC_NWG * (t + 1);
      while (__hip_atomic_load(counter, __ATOMIC_ACQUIRE,
                               __HIP_MEMORY_SCOPE_AGENT) < target)
        __builtin_amdgcn_s_sleep(1);
    }
    __syncthreads();
    // ---- async DMA: pull new h rows 0..7 (16 KB) from L2 into LDS ----
    {
      const uint4* src = (const uint4*)hglob;
      uint4* dst = (uint4*)h_s;
      for (int i = tid; i < RNN_B * RNN_H / 8; i += 64)
        async_load_b128(src + i, dst + i);
      wait_async0();
    }
    __syncthreads();
  }
}

// ---------------------------------------------------------------------------
// Launch
// ---------------------------------------------------------------------------
extern "C" void kernel_launch(void* const* d_in, const int* in_sizes, int n_in,
                              void* d_out, int out_size, void* d_ws,
                              size_t ws_size, hipStream_t stream) {
  (void)in_sizes; (void)n_in; (void)out_size; (void)ws_size;

  const int*   x    = (const int*)d_in[0];
  const float* emb  = (const float*)d_in[1];
  const float* Wxh  = (const float*)d_in[2];
  const float* Whh  = (const float*)d_in[3];
  const float* bh   = (const float*)d_in[4];
  const float* Why  = (const float*)d_in[5];
  const float* by   = (const float*)d_in[6];
  float*       y    = (float*)d_out;

  // workspace carve-up (256B aligned)
  char* w = (char*)d_ws;
  size_t off = 0;
  auto carve = [&](size_t bytes) {
    char* p = w + off;
    off = (off + bytes + 255) & ~(size_t)255;
    return p;
  };
  unsigned short* wxh16  = (unsigned short*)carve((size_t)RNN_E * RNN_H * 2);
  unsigned short* whh16  = (unsigned short*)carve((size_t)RNN_H * RNN_H * 2);
  unsigned short* why16  = (unsigned short*)carve((size_t)RNN_H * RNN_V * 2);
  float*          xh     = (float*)         carve((size_t)RNN_M * RNN_H * 4);
  unsigned short* hs16   = (unsigned short*)carve((size_t)RNN_M * RNN_H * 2);
  unsigned short* hglob  = (unsigned short*)carve((size_t)RNN_B * RNN_H * 2);
  int*            cnt    = (int*)           carve(256);

  // 1) weight conversions fp32 -> bf16
  cvt_f32_bf16_kernel<<<512, 256, 0, stream>>>(Wxh, wxh16, RNN_E * RNN_H);
  cvt_f32_bf16_kernel<<<1024, 256, 0, stream>>>(Whh, whh16, RNN_H * RNN_H);
  cvt_f32_bf16_kernel<<<4096, 256, 0, stream>>>(Why, why16, RNN_H * RNN_V);

  // 2) xh = emb[x] @ W_xh + b_h    (gathered-A bf16 WMMA GEMM)
  gemm_bf16_wmma_kernel<true><<<dim3(RNN_M / 128, RNN_H / 128), 256, 0, stream>>>(
      (const void*)emb, x, wxh16, bh, xh, RNN_M, RNN_H, RNN_E);

  // 3) recurrence init + persistent scan
  rnn_init_kernel<<<(RNN_B * RNN_H + 255) / 256, 256, 0, stream>>>(hglob, cnt);
  const size_t rec_lds = (size_t)16 * RNN_H * 2      // W_hh slice
                       + (size_t)16 * RNN_H * 2      // padded h tile
                       + 32 * 8 * sizeof(float);     // reduce buffer
  rnn_recurrence_kernel<<<REC_NWG, 64, rec_lds, stream>>>(whh16, xh, hs16,
                                                          hglob, cnt);

  // 4) y = hs @ W_hy + b_y   (dominant GEMM: 2048 x 32000 x 1024)
  gemm_bf16_wmma_kernel<false><<<dim3(RNN_M / 128, RNN_V / 128), 256, 0, stream>>>(
      (const void*)hs16, nullptr, why16, by, y, RNN_M, RNN_V, RNN_H);
}
